// SHNearfieldLightDecorator_14379550507349
// MI455X (gfx1250) — matrix operand order repaired and checked by reading, not compiled, not run on hardware
//
#include <hip/hip_runtime.h>
#include <hip/hip_bf16.h>

typedef __attribute__((ext_vector_type(16))) _Float16 v16h;
typedef __attribute__((ext_vector_type(8)))  float    v8f;

#define NGRID 128
#define NDIR  128
#define NB    25
#define PTS   (NGRID * NGRID * NGRID)
#define WAVES_PER_BLOCK 4
#define PTS_PER_WAVE 16

// Fast acos (Abramowitz-Stegun 4.4.45): |err| < 7e-5 rad, plenty for the
// exp(-16*acos^2) lobe. 3 FMAs + v_sqrt + select vs ~25-op libm path.
__device__ __forceinline__ float fast_acos(float x) {
    float ax = __builtin_fabsf(x);
    float r  = __builtin_amdgcn_sqrtf(1.0f - ax);
    float p  = __builtin_fmaf(ax, -0.0187293f, 0.0742610f);
    p        = __builtin_fmaf(ax, p, -0.2121144f);
    p        = __builtin_fmaf(ax, p, 1.5707288f);
    float v  = r * p;
    return (x >= 0.0f) ? v : (3.14159265f - v);
}

// Raw-hardware ln / gaussian-exp (args are bounded away from denorms here).
#define LN2F      0.6931471805599453f
#define NEG16LOG2E (-23.083120654223414f)   // -16 * log2(e)

__global__ __launch_bounds__(WAVES_PER_BLOCK * 32)
void sh_nearfield_wmma_kernel(const float* __restrict__ bases,
                              const float* __restrict__ th,
                              const float* __restrict__ ph,
                              const int*   __restrict__ iternum,
                              float* __restrict__ out)
{
    // Direction constants shared by all waves in the block.
    __shared__ float svx[NDIR], svy[NDIR], svz[NDIR], ssn[NDIR];
    // Per-wave raw l0 tile: [K=128 directions][N=16 points], f32.
    __shared__ float tile[WAVES_PER_BLOCK][NDIR][16];

    const int tid  = threadIdx.x;
    const int lane = tid & 31;
    const int wave = tid >> 5;
    const int n    = lane & 15;   // column (point) / A-row select
    const int h    = lane >> 4;   // half-wave select

    // ---- Phase 0a: direction grid -> LDS -------------------------------
    if (tid < NDIR) {
        float t = th[tid], p = ph[tid];
        float st = __sinf(t);
        svx[tid] = st * __sinf(p);
        svy[tid] = __cosf(t);
        svz[tid] = st * __cosf(p);
        ssn[tid] = st;
    }

    // ---- Phase 0b: A fragments (bases, f32 -> f16) ---------------------
    // ISA 16-bit A 16x32 layout: lane%16 = M; lanes 0-15 hold K {0..7,16..23},
    // lanes 16-31 hold K {8..15,24..31}; two packed f16 per VGPR.
    v16h afrag[2][4];
#pragma unroll
    for (int mt = 0; mt < 2; ++mt) {
        const int M = mt * 16 + n;
#pragma unroll
        for (int kc = 0; kc < 4; ++kc) {
            v16h a;
#pragma unroll
            for (int vv = 0; vv < 8; ++vv) {
                int k = kc * 32 + ((vv >> 2) * 16) + h * 8 + (vv & 3) * 2;
                float f0 = (M < NB) ? bases[M * NDIR + k]     : 0.0f;
                float f1 = (M < NB) ? bases[M * NDIR + k + 1] : 0.0f;
                a[2 * vv]     = (_Float16)f0;
                a[2 * vv + 1] = (_Float16)f1;
            }
            afrag[mt][kc] = a;
        }
    }

    __syncthreads();

    // ---- Light position (B == 1, b == 0) -------------------------------
    const float ph0 = (float)iternum[0] * (2.0f * 3.1415926f / 131.0f);
    const float lx = __sinf(ph0) * 0.3f;
    const float ly = 0.0f;
    const float lz = 0.4f;

    // ---- This lane's point (column n of the wave's 16-point tile) ------
    const int p0 = (blockIdx.x * WAVES_PER_BLOCK + wave) * PTS_PER_WAVE;
    const int p  = p0 + n;
    const int xi = p & (NGRID - 1);
    const int yi = (p >> 7) & (NGRID - 1);
    const int zi = p >> 14;
    const float step = 2.0f / (float)(NGRID - 1);
    const float fx = -1.0f + step * (float)xi;
    const float fy = -1.0f + step * (float)yi;
    const float fz = -1.0f + step * (float)zi;
    const float dx = lx - fx, dy = ly - fy, dz = lz - fz;
    const float norm2 = 1e-4f + dx * dx + dy * dy + dz * dz;
    const float rn = __builtin_amdgcn_rsqf(norm2);
    const float nx = dx * rn, ny = dy * rn, nz = dz * rn;

    // ---- Phase 1: raw l0 tile + column partial sums --------------------
    float (*mytile)[16] = tile[wave];
    float psum = 0.0f;
#pragma unroll 4
    for (int t = 0; t < 64; ++t) {
        int d = 2 * t + h;                       // conflict-free LDS stores
        float dp = svx[d] * nx + svy[d] * ny + svz[d] * nz;
        dp = fminf(0.999f, fmaxf(-0.999f, dp));
        float ac = fast_acos(dp);
        // exp(-16*ac^2) via raw v_exp_f32 (exp2)
        float v = __builtin_amdgcn_exp2f(NEG16LOG2E * ac * ac) * ssn[d];
        mytile[d][n] = v;
        psum += v;
    }
    const float sum = psum + __shfl_xor(psum, 16, 32);  // combine the 2 halves
    const float cs  = 4.0f / (sum * norm2);             // 4/(sum*norm^2)

    // ---- Phase 2: log-transform, pack f16 B fragments, WMMA ------------
    // B 32x16 layout: lane%16 = N; lanes 0-15 hold K {0..7,16..23}, lanes 16-31
    // hold K {8..15,24..31} of the chunk; two packed f16 per VGPR.
    v8f acc0 = {};
    v8f acc1 = {};
#pragma unroll
    for (int kc = 0; kc < 4; ++kc) {
        v16h b;
#pragma unroll
        for (int vv = 0; vv < 8; ++vv) {
            int d = kc * 32 + ((vv >> 2) * 16) + h * 8 + (vv & 3) * 2;
            // ln(1e-4 + y) via raw v_log_f32 (log2) * ln2; arg >= 1e-4.
            float f0 = __builtin_amdgcn_logf(1e-4f + mytile[d][n]     * cs) * LN2F;
            float f1 = __builtin_amdgcn_logf(1e-4f + mytile[d + 1][n] * cs) * LN2F;
            b[2 * vv]     = (_Float16)f0;
            b[2 * vv + 1] = (_Float16)f1;
        }
        acc0 = __builtin_amdgcn_wmma_f32_16x16x32_f16(
            false, afrag[0][kc], false, b, (short)0, acc0, false, false);
        acc1 = __builtin_amdgcn_wmma_f32_16x16x32_f16(
            false, afrag[1][kc], false, b, (short)0, acc1, false, false);
    }

    // ---- Store: C/D layout (VGPR r -> M = r + 8h, lane%16 -> N) --------
    const float scale = 9.869604401089358f / 128.0f;    // pi^2 / (2*L*L)
    const size_t P = (size_t)PTS;
    const size_t col = (size_t)(p0 + n);
#pragma unroll
    for (int r = 0; r < 8; ++r) {
        int M0 = r + 8 * h;
        float v0 = acc0[r] * scale;
        size_t base0 = (size_t)(M0 * 3) * P + col;
        out[base0]         = v0;
        out[base0 + P]     = v0;
        out[base0 + 2 * P] = v0;
        int M1 = 16 + r + 8 * h;
        if (M1 < NB) {
            float v1 = acc1[r] * scale;
            size_t base1 = (size_t)(M1 * 3) * P + col;
            out[base1]         = v1;
            out[base1 + P]     = v1;
            out[base1 + 2 * P] = v1;
        }
    }
}

extern "C" void kernel_launch(void* const* d_in, const int* in_sizes, int n_in,
                              void* d_out, int out_size, void* d_ws, size_t ws_size,
                              hipStream_t stream) {
    const float* bases   = (const float*)d_in[0];  // [25*128] f32
    const float* th      = (const float*)d_in[1];  // [128] f32
    const float* ph      = (const float*)d_in[2];  // [128] f32
    const int*   iternum = (const int*)d_in[3];    // scalar
    // d_in[4] (lightid) only defines B=1; value unused.
    float* out = (float*)d_out;                    // [75 * 128^3] f32

    const int blocks = PTS / (WAVES_PER_BLOCK * PTS_PER_WAVE);  // 32768
    sh_nearfield_wmma_kernel<<<blocks, WAVES_PER_BLOCK * 32, 0, stream>>>(
        bases, th, ph, iternum, out);
}
